// PoseRefineHeadSparseConv_40750649704652
// MI455X (gfx1250) — matrix operand order repaired and checked by reading, not compiled, not run on hardware
//
#include <hip/hip_runtime.h>
#include <hip/hip_bf16.h>

#define NPTS 262144
#define C64  64
#define G32  32
#define NSEG (32*32*32)
#define VOXSZ 0.1f
#define EPSV 1e-5f

typedef __attribute__((ext_vector_type(16))) __bf16 v16bf;
typedef __attribute__((ext_vector_type(8)))  float  v8f;

union Frag { v16bf v; unsigned u[8]; };

__device__ __forceinline__ unsigned short f2bf(float f) {
    unsigned u = __float_as_uint(f);
    unsigned r = u + 0x7FFFu + ((u >> 16) & 1u);   // round-to-nearest-even
    return (unsigned short)(r >> 16);
}

// ---------------------------------------------------------------- min coords
__global__ void k_min_coords(const float* __restrict__ pts, int* __restrict__ minc) {
    int i = blockIdx.x * blockDim.x + threadIdx.x;
    if (i >= NPTS) return;
    const float* P = pts + (size_t)i * 3;
    atomicMin(&minc[0], (int)floorf(P[0] / VOXSZ));
    atomicMin(&minc[1], (int)floorf(P[1] / VOXSZ));
    atomicMin(&minc[2], (int)floorf(P[2] / VOXSZ));
}

// ------------------------------------- encode (Linear+LN+ReLU) + scatter-add
__global__ void k_encode_voxelize(const float* __restrict__ pts,
                                  const float* __restrict__ w,
                                  const float* __restrict__ bias,
                                  const float* __restrict__ g,
                                  const float* __restrict__ beta,
                                  const int* __restrict__ minc,
                                  float* __restrict__ fsum,
                                  float* __restrict__ cnt) {
    int i = blockIdx.x * blockDim.x + threadIdx.x;
    if (i >= NPTS) return;
    const float* P = pts + (size_t)i * 3;
    float px = P[0], py = P[1], pz = P[2];
    // pass 1: LayerNorm stats over 64 channels (recompute, no spill)
    float s1 = 0.f, s2 = 0.f;
    #pragma unroll 8
    for (int c = 0; c < C64; ++c) {
        float h = fmaf(px, w[c], fmaf(py, w[64 + c], fmaf(pz, w[128 + c], bias[c])));
        s1 += h; s2 += h * h;
    }
    float mu   = s1 * (1.f / 64.f);
    float var  = s2 * (1.f / 64.f) - mu * mu;
    float rinv = rsqrtf(var + EPSV);
    int cx = min(max((int)floorf(px / VOXSZ) - minc[0], 0), G32 - 1);
    int cy = min(max((int)floorf(py / VOXSZ) - minc[1], 0), G32 - 1);
    int cz = min(max((int)floorf(pz / VOXSZ) - minc[2], 0), G32 - 1);
    size_t seg = ((size_t)cx * G32 + cy) * G32 + cz;
    float* dst = fsum + seg * C64;
    // pass 2: normalize + ReLU + scatter
    #pragma unroll 8
    for (int c = 0; c < C64; ++c) {
        float h = fmaf(px, w[c], fmaf(py, w[64 + c], fmaf(pz, w[128 + c], bias[c])));
        float v = fmaxf(0.f, fmaf((h - mu) * rinv, g[c], beta[c]));
        atomicAdd(&dst[c], v);
    }
    atomicAdd(&cnt[seg], 1.f);
}

// --------------------------------------------------- scatter-mean + bf16 copy
__global__ void k_voxel_mean(float* __restrict__ x, const float* __restrict__ cnt,
                             unsigned short* __restrict__ xbf, int* __restrict__ nact) {
    int i = blockIdx.x * blockDim.x + threadIdx.x;
    if (i >= NSEG * C64) return;
    int v = i >> 6;
    float c = cnt[v];
    float val = x[i] / fmaxf(c, 1.f);
    x[i] = val;
    xbf[i] = f2bf(val);
    if ((i & 63) == 0 && c > 0.f) atomicAdd(nact, 1);
}

// -------------------------- weight convert + transpose to [tap][out][in] bf16
__global__ void k_weight_cvt(const float* __restrict__ w, unsigned short* __restrict__ wt) {
    int i = blockIdx.x * blockDim.x + threadIdx.x;
    if (i >= 27 * C64 * C64) return;
    int o = i >> 12, rem = i & 4095, n = rem >> 6, k = rem & 63;
    wt[i] = f2bf(w[(o << 12) + (k << 6) + n]);    // wt[o][n][k] = w[o][k][n]
}

// ------------------------------------- SubMConv3d 3x3x3 64->64 via WMMA bf16
// One wave32 per 16-voxel z-strip x 64 out-channels. 27 taps x 2 k-steps x
// 4 n-tiles = 216 v_wmma_f32_16x16x32_bf16 per wave, f32 accumulation.
// Z-boundary handled by indexing a zero guard row (voxel NSEG), so all
// fragment loads are unconditional b128s and EXEC stays all-ones throughout.
__global__ void __launch_bounds__(128)
k_subm_conv(const unsigned* __restrict__ xbf32,   // (NSEG+1) x 32 uints; row NSEG = 0
            const unsigned* __restrict__ wbf32,   // 27*64 x 32 uints (K-major)
            float* __restrict__ y) {
    int lane = threadIdx.x & 31;
    // tile id is wave-uniform: force it into an SGPR so bounds tests below
    // become s_cmp/s_cbranch (no EXEC manipulation around the WMMAs).
    int tile = __builtin_amdgcn_readfirstlane(blockIdx.x * 4 + (threadIdx.x >> 5));
    int zt = tile & 1;
    int cy = (tile >> 1) & 31;
    int cx = (tile >> 6);
    int z0 = zt << 4;
    int row = lane & 15;                          // M for A, N for B, col for D
    int hi  = lane >> 4;

    __builtin_prefetch(wbf32, 0, 3);              // global_prefetch_b8 of weight tile

    v8f zero = {0.f, 0.f, 0.f, 0.f, 0.f, 0.f, 0.f, 0.f};
    v8f acc[4] = {zero, zero, zero, zero};

    for (int dx = -1; dx <= 1; ++dx) {
        int nx = cx + dx; if ((unsigned)nx >= 32u) continue;      // scalar branch
        for (int dy = -1; dy <= 1; ++dy) {
            int ny = cy + dy; if ((unsigned)ny >= 32u) continue;  // scalar branch
            int xybase = ((nx << 5) + ny) << 5;                   // SGPR
            for (int dz = -1; dz <= 1; ++dz) {
                int nz = z0 + row + dz;                            // per-lane
                bool valid = (unsigned)nz < 32u;
                int seg = valid ? (xybase + nz) : NSEG;            // guard row if OOB
                const unsigned* arow = xbf32 + (size_t)seg * 32;
                int o = ((dx + 1) * 3 + (dy + 1)) * 3 + (dz + 1);
                const unsigned* wtap = wbf32 + (size_t)o * 64 * 32;
                #pragma unroll
                for (int ks = 0; ks < 2; ++ks) {
                    // A fragment (16x32 bf16): ISA 7.12.2 16-bit A layout
                    Frag a;
                    #pragma unroll
                    for (int j = 0; j < 4; ++j)
                        a.u[j] = arow[ks * 16 + hi * 4 + j];       // -> b128
                    #pragma unroll
                    for (int j = 4; j < 8; ++j)
                        a.u[j] = arow[ks * 16 + 8 + hi * 4 + (j - 4)]; // -> b128
                    #pragma unroll
                    for (int nt = 0; nt < 4; ++nt) {
                        // B fragment (32x16 bf16): lane = out-channel, K contiguous
                        const unsigned* brow = wtap + (size_t)(nt * 16 + row) * 32;
                        Frag b;
                        #pragma unroll
                        for (int j = 0; j < 8; ++j)
                            b.u[j] = brow[ks * 16 + (j >> 2) * 8 + hi * 4 + (j & 3)];   // -> 2x b128
                        acc[nt] = __builtin_amdgcn_wmma_f32_16x16x32_bf16(
                            false, a.v, false, b.v, (short)0, acc[nt], false, false);
                    }
                }
            }
        }
    }
    // D layout: VGPR r -> M = r + hi*8, N = lane&15
    size_t base = (((size_t)cx << 5) + cy) * 32 + z0;
    #pragma unroll
    for (int r = 0; r < 8; ++r) {
        int m = r + hi * 8;
        float* yr = y + (base + m) * C64 + row;
        yr[0]  = acc[0][r];
        yr[16] = acc[1][r];
        yr[32] = acc[2][r];
        yr[48] = acc[3][r];
    }
}

// ------------------------------------------------ BN batch stats (active only)
__global__ void k_bn_stats(const float* __restrict__ y, const float* __restrict__ cnt,
                           float* __restrict__ stats) {
    int c = threadIdx.x;                  // 64 threads, one channel each
    int v0 = blockIdx.x * 128;            // 256 blocks * 128 voxels = NSEG
    float s = 0.f, q = 0.f;
    for (int v = v0; v < v0 + 128; ++v) {
        float m = cnt[v] > 0.f ? 1.f : 0.f;
        float val = y[(size_t)v * C64 + c] * m;
        s += val; q += val * val;
    }
    atomicAdd(&stats[c], s);
    atomicAdd(&stats[64 + c], q);
}

// ---------------------------------------------- BN + ReLU (+ bf16 for conv2)
__global__ void k_bn_relu(const float* __restrict__ y, const float* __restrict__ cnt,
                          const float* __restrict__ stats, const int* __restrict__ nact,
                          const float* __restrict__ g, const float* __restrict__ b,
                          float* __restrict__ out, unsigned short* __restrict__ outbf) {
    int i = blockIdx.x * blockDim.x + threadIdx.x;
    if (i >= NSEG * C64) return;
    int v = i >> 6, c = i & 63;
    float n    = fmaxf((float)*nact, 1.f);
    float mu   = stats[c] / n;
    float var  = stats[64 + c] / n - mu * mu;
    float rinv = rsqrtf(var + EPSV);
    float m    = cnt[v] > 0.f ? 1.f : 0.f;
    float val  = fmaxf(0.f, fmaf((y[i] - mu) * rinv, g[c], b[c]) * m);
    out[i]   = val;
    outbf[i] = f2bf(val);
}

// ------------------------------------- BN + residual + ReLU (second BN block)
__global__ void k_bn_res_relu(const float* __restrict__ y, const float* __restrict__ cnt,
                              const float* __restrict__ stats, const int* __restrict__ nact,
                              const float* __restrict__ g, const float* __restrict__ b,
                              const float* __restrict__ xres, float* __restrict__ out) {
    int i = blockIdx.x * blockDim.x + threadIdx.x;
    if (i >= NSEG * C64) return;
    int v = i >> 6, c = i & 63;
    float n    = fmaxf((float)*nact, 1.f);
    float mu   = stats[c] / n;
    float var  = stats[64 + c] / n - mu * mu;
    float rinv = rsqrtf(var + EPSV);
    float m    = cnt[v] > 0.f ? 1.f : 0.f;
    out[i] = fmaxf(0.f, fmaf((y[i] - mu) * rinv, g[c], b[c]) * m + xres[i]);
}

// -------------------------- masked channel max (post-ReLU values are >= 0)
__global__ void k_chan_max(const float* __restrict__ r, const float* __restrict__ cnt,
                           float* __restrict__ bmax) {
    int c = threadIdx.x;                   // 64 channels
    float mx = 0.f;
    for (int v = blockIdx.x; v < NSEG; v += gridDim.x) {
        float m = cnt[v] > 0.f ? 1.f : 0.f;
        mx = fmaxf(mx, r[(size_t)v * C64 + c] * m);
    }
    atomicMax((unsigned*)&bmax[c], __float_as_uint(mx));   // valid for floats >= 0
}

// ------------------------------------------------------------- MLP head (6)
__global__ void k_head(const float* __restrict__ bmax,
                       const float* __restrict__ w1, const float* __restrict__ b1,
                       const float* __restrict__ w2, const float* __restrict__ b2,
                       float* __restrict__ out) {
    __shared__ float comb[64];
    __shared__ float h[64];
    int t = threadIdx.x;
    comb[t] = bmax[t] + bmax[64 + t];
    __syncthreads();
    float a = b1[t];
    #pragma unroll 8
    for (int i = 0; i < 64; ++i) a = fmaf(comb[i], w1[i * 64 + t], a);
    h[t] = fmaxf(a, 0.f);
    __syncthreads();
    if (t < 6) {
        float a2 = b2[t];
        #pragma unroll 8
        for (int i = 0; i < 64; ++i) a2 = fmaf(h[i], w2[i * 6 + t], a2);
        out[t] = a2;
    }
}

extern "C" void kernel_launch(void* const* d_in, const int* in_sizes, int n_in,
                              void* d_out, int out_size, void* d_ws, size_t ws_size,
                              hipStream_t stream) {
    (void)in_sizes; (void)n_in; (void)out_size; (void)ws_size;
    const float* src     = (const float*)d_in[0];
    const float* tgt     = (const float*)d_in[1];
    const float* enc_w   = (const float*)d_in[2];
    const float* enc_b   = (const float*)d_in[3];
    const float* ln_g    = (const float*)d_in[4];
    const float* ln_b    = (const float*)d_in[5];
    const float* conv1_w = (const float*)d_in[6];
    const float* bn1_g   = (const float*)d_in[7];
    const float* bn1_b   = (const float*)d_in[8];
    const float* conv2_w = (const float*)d_in[9];
    const float* bn2_g   = (const float*)d_in[10];
    const float* bn2_b   = (const float*)d_in[11];
    const float* hw1     = (const float*)d_in[12];
    const float* hb1     = (const float*)d_in[13];
    const float* hw2     = (const float*)d_in[14];
    const float* hb2     = (const float*)d_in[15];

    char* ws = (char*)d_ws;
    size_t p = 0;
    auto alloc = [&](size_t n) { size_t r = p; p = (p + n + 255) & ~(size_t)255; return r; };
    size_t o_minc  = alloc(16);
    size_t o_nact  = alloc(16);
    size_t o_stats = alloc(128 * 4);
    size_t o_bmax  = alloc(128 * 4);
    size_t o_wbf1  = alloc((size_t)27 * 64 * 64 * 2);
    size_t o_wbf2  = alloc((size_t)27 * 64 * 64 * 2);
    size_t o_cnt   = alloc((size_t)NSEG * 4);
    size_t o_x     = alloc((size_t)NSEG * 64 * 4);         // fsum, then voxel means
    size_t o_xbf   = alloc((size_t)(NSEG + 1) * 64 * 2);   // +1 zero guard row
    size_t o_y     = alloc((size_t)NSEG * 64 * 4);         // conv output (reused)
    size_t o_a1    = alloc((size_t)NSEG * 64 * 4);         // activations / result
    size_t o_a1bf  = alloc((size_t)(NSEG + 1) * 64 * 2);   // +1 zero guard row

    int*            minc  = (int*)(ws + o_minc);
    int*            nact  = (int*)(ws + o_nact);
    float*          stats = (float*)(ws + o_stats);
    float*          bmax  = (float*)(ws + o_bmax);
    unsigned short* wbf1  = (unsigned short*)(ws + o_wbf1);
    unsigned short* wbf2  = (unsigned short*)(ws + o_wbf2);
    float*          cnt   = (float*)(ws + o_cnt);
    float*          xf    = (float*)(ws + o_x);
    unsigned short* xbf   = (unsigned short*)(ws + o_xbf);
    float*          yf    = (float*)(ws + o_y);
    float*          a1f   = (float*)(ws + o_a1);
    unsigned short* a1bf  = (unsigned short*)(ws + o_a1bf);

    // weights -> bf16, K-major (shared by both branches)
    k_weight_cvt<<<(27 * 64 * 64 + 255) / 256, 256, 0, stream>>>(conv1_w, wbf1);
    k_weight_cvt<<<(27 * 64 * 64 + 255) / 256, 256, 0, stream>>>(conv2_w, wbf2);
    hipMemsetAsync(bmax, 0, 128 * 4, stream);
    // zero guard rows (row NSEG) — never written by kernels
    hipMemsetAsync(xbf  + (size_t)NSEG * 64, 0, 64 * 2, stream);
    hipMemsetAsync(a1bf + (size_t)NSEG * 64, 0, 64 * 2, stream);

    for (int b = 0; b < 2; ++b) {
        const float* pts = b ? tgt : src;
        hipMemsetAsync(minc, 0x7F, 16, stream);                 // ~INT_MAX init
        hipMemsetAsync(nact, 0, 16, stream);
        hipMemsetAsync(stats, 0, 128 * 4, stream);
        hipMemsetAsync(cnt, 0, (size_t)NSEG * 4, stream);
        hipMemsetAsync(xf, 0, (size_t)NSEG * 64 * 4, stream);

        k_min_coords<<<NPTS / 256, 256, 0, stream>>>(pts, minc);
        k_encode_voxelize<<<NPTS / 256, 256, 0, stream>>>(pts, enc_w, enc_b, ln_g, ln_b,
                                                          minc, xf, cnt);
        k_voxel_mean<<<NSEG * 64 / 256, 256, 0, stream>>>(xf, cnt, xbf, nact);

        k_subm_conv<<<512, 128, 0, stream>>>((const unsigned*)xbf, (const unsigned*)wbf1, yf);
        k_bn_stats<<<256, 64, 0, stream>>>(yf, cnt, stats);
        k_bn_relu<<<NSEG * 64 / 256, 256, 0, stream>>>(yf, cnt, stats, nact, bn1_g, bn1_b,
                                                       a1f, a1bf);

        k_subm_conv<<<512, 128, 0, stream>>>((const unsigned*)a1bf, (const unsigned*)wbf2, yf);
        hipMemsetAsync(stats, 0, 128 * 4, stream);
        k_bn_stats<<<256, 64, 0, stream>>>(yf, cnt, stats);
        k_bn_res_relu<<<NSEG * 64 / 256, 256, 0, stream>>>(yf, cnt, stats, nact, bn2_g, bn2_b,
                                                           xf, a1f);

        k_chan_max<<<64, 64, 0, stream>>>(a1f, cnt, bmax + b * 64);
    }

    k_head<<<1, 64, 0, stream>>>(bmax, hw1, hb1, hw2, hb2, (float*)d_out);
}
